// GraphTransformer_3375844295125
// MI455X (gfx1250) — matrix operand order repaired and checked by reading, not compiled, not run on hardware
//
#include <hip/hip_runtime.h>
#include <hip/hip_bf16.h>

// ---------------------------------------------------------------------------
// CDNA5 (gfx1250) structure-aware transformer encoder.
// - Dense matmuls + both relation einsums via V_WMMA_F32_16X16X32_BF16.
// - Fragment-ready LDS layouts: per-lane fragments are 32 contiguous bytes
//   -> 2x ds_load_b128 per fragment.
// - Vectorized staging: global_load_b128 / b32 + v_cvt_pk_bf16 -> ds_store_b64.
// - GEMM templated on (N,K,RELU,RESID): constant-folded addressing, no spills.
// ---------------------------------------------------------------------------

#define DMODEL 512
#define NHEAD  8
#define DKH    64
#define DFFN   1024
#define NLAY   4
#define BATCH  8
#define SEQ    256
#define MROWS  (BATCH*SEQ)   // 2048
#define NRV    128           // distinct relation ids

typedef __attribute__((ext_vector_type(16))) __bf16        v16bf;
typedef __attribute__((ext_vector_type(4)))  __bf16        v4bf;
typedef __attribute__((ext_vector_type(8)))  float         v8f;
typedef __attribute__((ext_vector_type(8)))  unsigned int  v8u;

__device__ __forceinline__ __bf16 f2bf(float x) { return (__bf16)x; }

__device__ __forceinline__ v8f wmma_bf16(v16bf a, v16bf b, v8f c) {
    return __builtin_amdgcn_wmma_f32_16x16x32_bf16(
        false, a, false, b, (short)0, c, false, false);
}

// Load a 32-byte fragment (16 bf16) from LDS: lowers to 2x ds_load_b128.
__device__ __forceinline__ v16bf ld_frag(const __bf16* p) {
    v8u r = *(const v8u*)p;
    return __builtin_bit_cast(v16bf, r);
}

__device__ __forceinline__ v4bf pack4(float a, float b, float c, float d) {
    v4bf r; r[0] = f2bf(a); r[1] = f2bf(b); r[2] = f2bf(c); r[3] = f2bf(d);
    return r;
}

// ---------------------------------------------------------------------------
// Embedding + sinusoidal positional encoding.
// ---------------------------------------------------------------------------
__global__ void embed_pe_kernel(const int* __restrict__ ids,
                                const float* __restrict__ emb,
                                float* __restrict__ x) {
    int row = blockIdx.x;
    int pos = row % SEQ;
    int id  = ids[row];
    int t   = threadIdx.x;
    int d0  = 2 * t;
    float inv = __powf(10000.0f, -((float)d0) / (float)DMODEL);
    float ang = (float)pos * inv;
    x[(size_t)row * DMODEL + d0]     = emb[(size_t)id * DMODEL + d0]     + __sinf(ang);
    x[(size_t)row * DMODEL + d0 + 1] = emb[(size_t)id * DMODEL + d0 + 1] + __cosf(ang);
}

// ---------------------------------------------------------------------------
// LayerNorm over D=512. One block (256 threads, 8 wave32) per row.
// ---------------------------------------------------------------------------
__global__ void layernorm_kernel(const float* __restrict__ x,
                                 const float* __restrict__ g,
                                 const float* __restrict__ bta,
                                 float* __restrict__ out) {
    __shared__ float ssum[8];
    __shared__ float ssq[8];
    int row = blockIdx.x;
    int t   = threadIdx.x;
    float a = x[(size_t)row * DMODEL + t];
    float b = x[(size_t)row * DMODEL + t + 256];
    float s = a + b;
    float q = a * a + b * b;
    #pragma unroll
    for (int off = 1; off < 32; off <<= 1) {
        s += __shfl_xor(s, off, 32);
        q += __shfl_xor(q, off, 32);
    }
    if ((t & 31) == 0) { ssum[t >> 5] = s; ssq[t >> 5] = q; }
    __syncthreads();
    float S = 0.f, Q = 0.f;
    #pragma unroll
    for (int i = 0; i < 8; i++) { S += ssum[i]; Q += ssq[i]; }
    float m   = S * (1.0f / DMODEL);
    float var = Q * (1.0f / DMODEL) - m * m;
    float r   = rsqrtf(var + 1e-5f);
    out[(size_t)row * DMODEL + t]       = (a - m) * r * g[t]       + bta[t];
    out[(size_t)row * DMODEL + t + 256] = (b - m) * r * g[t + 256] + bta[t + 256];
}

// ---------------------------------------------------------------------------
// skT/svT tables: out[128,64] = rel_table[128,64] @ W[64,64] + bias   (tiny)
// ---------------------------------------------------------------------------
__global__ void reltable_kernel(const float* __restrict__ rt,
                                const float* __restrict__ Wm,
                                const float* __restrict__ bm,
                                float* __restrict__ o) {
    int idx = blockIdx.x * 256 + threadIdx.x;   // < 8192
    int rv = idx >> 6, dk = idx & 63;
    float s = bm[dk];
    #pragma unroll 8
    for (int j = 0; j < 64; j++) s += rt[rv * 64 + j] * Wm[j * 64 + dk];
    o[idx] = s;
}

// ---------------------------------------------------------------------------
// Dense GEMM: C[M,N] = A[M,K] @ W[K,N] + bias (+residual) (+relu)
// Block = 128 threads (4 waves). Tile M=128 (32 rows/wave), N=64, BK=32.
// Compile-time N,K -> constant-folded addressing.
// ---------------------------------------------------------------------------
template<int N, int K, bool RELU, bool HASRES>
__global__ void gemm_bf16_kernel(const float* __restrict__ A,
                                 const float* __restrict__ W,
                                 const float* __restrict__ bias,
                                 const float* __restrict__ resid,
                                 float* __restrict__ C) {
    __shared__ __align__(16) __bf16 AsF[8 * 512];   // 8 row-tiles, frag layout
    __shared__ __align__(16) __bf16 BsF[4 * 512];   // 4 col-tiles, frag layout
    const int n0   = blockIdx.x * 64;
    const int m0   = blockIdx.y * 128;
    const int tid  = threadIdx.x;
    const int lane = tid & 31;
    const int w    = tid >> 5;
    const int mr   = lane & 15;
    const int roff = (lane & 16) ? 8 : 0;

    v8f acc[2][4] = {};

    for (int k0 = 0; k0 < K; k0 += 32) {
        // ---- stage A: 128x32 fp32 -> bf16 frag layout (float4 -> ds_store_b64)
        #pragma unroll
        for (int g = 0; g < 8; g++) {
            int f   = g * 128 + tid;          // float4 index, 1024 total
            int ra  = f >> 3;                 // row 0..127
            int ca4 = (f & 7) * 4;            // K-col 0..28 step 4
            const float4 av = *(const float4*)&A[(size_t)(m0 + ra) * K + k0 + ca4];
            int la = ((ca4 & 16) ? 16 : 0) + (ra & 15);
            int ja = (ca4 & 15);
            *(v4bf*)&AsF[(ra >> 4) * 512 + la * 16 + ja] = pack4(av.x, av.y, av.z, av.w);
        }
        // ---- stage B: 32x64 fp32 -> bf16 frag layout (4 coalesced b32 -> b64)
        #pragma unroll
        for (int g = 0; g < 4; g++) {
            int cb  = tid & 63;
            int rb0 = (tid >> 6) * 4 + g * 8; // 0..28 step 4
            const float* wp = &W[(size_t)(k0 + rb0) * N + n0 + cb];
            float b0 = wp[0], b1 = wp[N], b2 = wp[2 * N], b3 = wp[3 * N];
            int lb = ((rb0 & 16) ? 16 : 0) + (cb & 15);
            *(v4bf*)&BsF[(cb >> 4) * 512 + lb * 16 + (rb0 & 15)] = pack4(b0, b1, b2, b3);
        }
        if (k0 + 32 < K)
            __builtin_prefetch(&W[(size_t)(k0 + 32 + (tid >> 6)) * N + n0 + (tid & 63)], 0, 1);
        __syncthreads();

        v16bf bfr[4];
        #pragma unroll
        for (int t = 0; t < 4; t++) bfr[t] = ld_frag(&BsF[t * 512 + lane * 16]);
        #pragma unroll
        for (int mA = 0; mA < 2; mA++) {
            v16bf af = ld_frag(&AsF[(w * 2 + mA) * 512 + lane * 16]);
            #pragma unroll
            for (int t = 0; t < 4; t++)
                acc[mA][t] = wmma_bf16(af, bfr[t], acc[mA][t]);
        }
        __syncthreads();
    }

    #pragma unroll
    for (int mA = 0; mA < 2; mA++) {
        #pragma unroll
        for (int t = 0; t < 4; t++) {
            #pragma unroll
            for (int r = 0; r < 8; r++) {
                int row = m0 + w * 32 + mA * 16 + r + roff;
                int col = n0 + t * 16 + mr;
                float val = acc[mA][t][r] + bias[col];
                if (HASRES) val += resid[(size_t)row * N + col];
                if (RELU)   val = fmaxf(val, 0.0f);
                C[(size_t)row * N + col] = val;
            }
        }
    }
}

// ---------------------------------------------------------------------------
// Structure-aware attention, one (b, h, 64-q-row tile) per block (4 waves).
// Phase 0: Qsk[64,128] = q_tile @ skT^T           (WMMA -> LDS aux)
// Phase 1: S = q.kT (WMMA) + Qsk[rel] gather, scale, mask
// Softmax in LDS (wave-private rows)
// Phase 2: Pagg[64,128] = bincount_rel(P)          (ds_add_f32 atomics)
//          ctx = P @ V (WMMA, V staged frag-ready) + Pagg @ svT (WMMA)
// ---------------------------------------------------------------------------
__global__ void attention_kernel(const float* __restrict__ q,
                                 const float* __restrict__ k,
                                 const float* __restrict__ v,
                                 const float* __restrict__ skT,
                                 const float* __restrict__ svT,
                                 const int* __restrict__ rel,
                                 const int* __restrict__ mask,
                                 float* __restrict__ ctx) {
    __shared__ float S[64 * SEQ];                    // 64 KB scores/probs
    __shared__ float aux[64 * NRV];                  // 32 KB Qsk then Pagg
    __shared__ __align__(16) __bf16 vF[32 * 512];    // 32 KB V frag-ready

    int bb = blockIdx.z, hh = blockIdx.y;
    int q0 = blockIdx.x * 64;
    int tid = threadIdx.x, lane = tid & 31, w = tid >> 5;
    int kbA = (lane & 16) ? 16 : 0;
    int kbB = (lane & 16) ? 16 : 0;
    int mr  = lane & 15;
    int roff = (lane & 16) ? 8 : 0;

    // ---- cooperative staging of V[256,64] into fragment-ready bf16 ----
    {
        const float* vbase = v + (size_t)bb * SEQ * DMODEL + hh * DKH;
        #pragma unroll
        for (int g = 0; g < 32; g++) {
            int d   = tid & 63;
            int kp0 = (tid >> 6) * 4 + g * 8;        // 0..252 step 4
            const float* vp = vbase + (size_t)kp0 * DMODEL + d;
            float x0 = vp[0], x1 = vp[DMODEL], x2 = vp[2 * DMODEL], x3 = vp[3 * DMODEL];
            int lane_d = ((kp0 & 16) ? 16 : 0) + (d & 15);
            int grp    = (kp0 >> 5) * 4 + (d >> 4);  // ks*4 + t
            *(v4bf*)&vF[grp * 512 + lane_d * 16 + (kp0 & 15)] = pack4(x0, x1, x2, x3);
        }
    }
    __syncthreads();

    // hoisted q A-fragments for the two K=32 steps over DK=64
    int qrowA = q0 + w * 16 + mr;
    const float* qrp = q + ((size_t)(bb * SEQ + qrowA) * DMODEL + hh * DKH);
    v16bf qa[2];
    #pragma unroll
    for (int kk = 0; kk < 2; kk++) {
        #pragma unroll
        for (int j = 0; j < 8; j++)  qa[kk][j] = f2bf(qrp[kk * 32 + kbA + j]);
        #pragma unroll
        for (int j = 8; j < 16; j++) qa[kk][j] = f2bf(qrp[kk * 32 + kbA + j]);
    }

    // ---- phase 0: Qsk = q_tile @ skT^T  -> aux[row][rv] ----
    #pragma unroll
    for (int t = 0; t < 8; t++) {
        v8f acc = {};
        #pragma unroll
        for (int kk = 0; kk < 2; kk++) {
            const float* sp = skT + (size_t)(t * 16 + mr) * DKH + kk * 32 + kbB;
            v16bf bf_;
            #pragma unroll
            for (int j = 0; j < 16; j++) bf_[j] = f2bf(sp[j]);
            acc = wmma_bf16(qa[kk], bf_, acc);
        }
        #pragma unroll
        for (int r = 0; r < 8; r++)
            aux[(w * 16 + r + roff) * NRV + t * 16 + mr] = acc[r];
    }

    // ---- phase 1: scores ----
    for (int kt = 0; kt < 16; kt++) {
        v8f s = {};
        int kc = kt * 16 + mr;
        const float* krp = k + ((size_t)(bb * SEQ + kc) * DMODEL + hh * DKH);
        #pragma unroll
        for (int kk = 0; kk < 2; kk++) {
            const float* kp = krp + kk * 32 + kbB;
            v16bf kf;
            #pragma unroll
            for (int j = 0; j < 16; j++) kf[j] = f2bf(kp[j]);
            s = wmma_bf16(qa[kk], kf, s);
        }
        int mk = mask[bb * SEQ + kc];
        #pragma unroll
        for (int r = 0; r < 8; r++) {
            int qr  = q0 + w * 16 + r + roff;
            int rid = rel[(size_t)(bb * SEQ + qr) * SEQ + kc];
            float val = (s[r] + aux[(w * 16 + r + roff) * NRV + rid]) * 0.125f;
            if (mk == 0) val = -1e9f;
            S[(w * 16 + r + roff) * SEQ + kc] = val;
        }
    }

    // ---- softmax: 2 lanes per row, 16 rows per wave (wave-private) ----
    {
        int rloc = w * 16 + (lane >> 1);
        int half = lane & 1;
        float mx = -3.4e38f;
        for (int kk2 = 0; kk2 < 128; kk2++)
            mx = fmaxf(mx, S[rloc * SEQ + half * 128 + kk2]);
        mx = fmaxf(mx, __shfl_xor(mx, 1, 32));
        float sum = 0.0f;
        for (int kk2 = 0; kk2 < 128; kk2++) {
            float e = __expf(S[rloc * SEQ + half * 128 + kk2] - mx);
            S[rloc * SEQ + half * 128 + kk2] = e;
            sum += e;
        }
        sum += __shfl_xor(sum, 1, 32);
        float inv = 1.0f / sum;
        for (int kk2 = 0; kk2 < 128; kk2++)
            S[rloc * SEQ + half * 128 + kk2] *= inv;
    }

    // ---- phase 2a: Pagg = weighted bincount of P over rel ids ----
    for (int i = lane; i < 16 * NRV; i += 32) aux[w * 16 * NRV + i] = 0.0f;
    {
        int rloc = w * 16 + (lane >> 1);
        int half = lane & 1;
        int qr = q0 + rloc;
        const int* rrp = rel + (size_t)(bb * SEQ + qr) * SEQ + half * 128;
        for (int kk2 = 0; kk2 < 128; kk2++) {
            float p = S[rloc * SEQ + half * 128 + kk2];
            atomicAdd(&aux[rloc * NRV + rrp[kk2]], p);
        }
    }

    // ---- phase 2b: ctx = P @ V  +  Pagg @ svT ----
    v8f cacc[4] = {};
    for (int ks = 0; ks < 8; ks++) {
        const float* prow = &S[(w * 16 + mr) * SEQ + ks * 32 + kbA];
        v16bf pa;
        #pragma unroll
        for (int j = 0; j < 8; j++)  pa[j] = f2bf(prow[j]);
        #pragma unroll
        for (int j = 8; j < 16; j++) pa[j] = f2bf(prow[j]);
        #pragma unroll
        for (int t = 0; t < 4; t++) {
            v16bf vbf = ld_frag(&vF[(ks * 4 + t) * 512 + lane * 16]);
            cacc[t] = wmma_bf16(pa, vbf, cacc[t]);
        }
    }
    #pragma unroll
    for (int ks = 0; ks < 4; ks++) {       // K = 128 rv bins
        const float* grow = &aux[(w * 16 + mr) * NRV + ks * 32 + kbA];
        v16bf pa2;
        #pragma unroll
        for (int j = 0; j < 8; j++)  pa2[j] = f2bf(grow[j]);
        #pragma unroll
        for (int j = 8; j < 16; j++) pa2[j] = f2bf(grow[j]);
        #pragma unroll
        for (int t = 0; t < 4; t++) {
            v16bf sb;
            #pragma unroll
            for (int j = 0; j < 16; j++)
                sb[j] = f2bf(svT[(size_t)(ks * 32 + kbB + j) * DKH + t * 16 + mr]);
            cacc[t] = wmma_bf16(pa2, sb, cacc[t]);
        }
    }

    #pragma unroll
    for (int t = 0; t < 4; t++) {
        #pragma unroll
        for (int r = 0; r < 8; r++) {
            int row = q0 + w * 16 + r + roff;
            int d   = t * 16 + mr;
            ctx[(size_t)(bb * SEQ + row) * DMODEL + hh * DKH + d] = cacc[t][r];
        }
    }
}

// ---------------------------------------------------------------------------
// Host orchestration
// ---------------------------------------------------------------------------
extern "C" void kernel_launch(void* const* d_in, const int* in_sizes, int n_in,
                              void* d_out, int out_size, void* d_ws, size_t ws_size,
                              hipStream_t stream) {
    const int*   con    = (const int*)d_in[0];
    const int*   conm   = (const int*)d_in[1];
    const int*   relS   = (const int*)d_in[2];
    const int*   tcon   = (const int*)d_in[3];
    const int*   tconm  = (const int*)d_in[4];
    const int*   relT   = (const int*)d_in[5];
    const float* emb    = (const float*)d_in[6];
    const float* rtab   = (const float*)d_in[7];
    const float* Wq     = (const float*)d_in[8];
    const float* Wk     = (const float*)d_in[9];
    const float* Wv     = (const float*)d_in[10];
    const float* Wo     = (const float*)d_in[11];
    const float* bq     = (const float*)d_in[12];
    const float* bk     = (const float*)d_in[13];
    const float* bv     = (const float*)d_in[14];
    const float* bo     = (const float*)d_in[15];
    const float* Wsk    = (const float*)d_in[16];
    const float* Wsv    = (const float*)d_in[17];
    const float* bsk    = (const float*)d_in[18];
    const float* bsv    = (const float*)d_in[19];
    const float* W1     = (const float*)d_in[20];
    const float* b1     = (const float*)d_in[21];
    const float* W2     = (const float*)d_in[22];
    const float* b2     = (const float*)d_in[23];
    const float* ln1g   = (const float*)d_in[24];
    const float* ln1b   = (const float*)d_in[25];
    const float* ln2g   = (const float*)d_in[26];
    const float* ln2b   = (const float*)d_in[27];
    const float* lnfg   = (const float*)d_in[28];
    const float* lnfb   = (const float*)d_in[29];

    const size_t XSZ = (size_t)MROWS * DMODEL;
    float* x    = (float*)d_ws;
    float* h    = x    + XSZ;
    float* qb   = h    + XSZ;
    float* kb   = qb   + XSZ;
    float* vb   = kb   + XSZ;
    float* cx   = vb   + XSZ;
    float* ffn  = cx   + XSZ;                      // 2048 x 1024
    float* skT  = ffn  + (size_t)MROWS * DFFN;
    float* svT  = skT  + NRV * DKH;

    dim3 gemmBlk(128);
    dim3 gD(DMODEL / 64, MROWS / 128);
    dim3 gF(DFFN   / 64, MROWS / 128);
    dim3 attnGrid(SEQ / 64, NHEAD, BATCH);

    for (int p = 0; p < 2; p++) {
        const int* ids = p ? tcon  : con;
        const int* msk = p ? tconm : conm;
        const int* rl  = p ? relT  : relS;
        float* outp = (float*)d_out + (size_t)p * XSZ;

        embed_pe_kernel<<<MROWS, 256, 0, stream>>>(ids, emb, x);

        for (int i = 0; i < NLAY; i++) {
            const float* Wq_i  = Wq  + (size_t)i * DMODEL * DMODEL;
            const float* Wk_i  = Wk  + (size_t)i * DMODEL * DMODEL;
            const float* Wv_i  = Wv  + (size_t)i * DMODEL * DMODEL;
            const float* Wo_i  = Wo  + (size_t)i * DMODEL * DMODEL;
            const float* W1_i  = W1  + (size_t)i * DMODEL * DFFN;
            const float* W2_i  = W2  + (size_t)i * DFFN * DMODEL;

            layernorm_kernel<<<MROWS, 256, 0, stream>>>(x, ln1g + i * DMODEL, ln1b + i * DMODEL, h);

            gemm_bf16_kernel<DMODEL, DMODEL, false, false>
                <<<gD, gemmBlk, 0, stream>>>(h, Wq_i, bq + i * DMODEL, nullptr, qb);
            gemm_bf16_kernel<DMODEL, DMODEL, false, false>
                <<<gD, gemmBlk, 0, stream>>>(h, Wk_i, bk + i * DMODEL, nullptr, kb);
            gemm_bf16_kernel<DMODEL, DMODEL, false, false>
                <<<gD, gemmBlk, 0, stream>>>(h, Wv_i, bv + i * DMODEL, nullptr, vb);

            reltable_kernel<<<32, 256, 0, stream>>>(rtab, Wsk + (size_t)i * DKH * DKH, bsk + i * DKH, skT);
            reltable_kernel<<<32, 256, 0, stream>>>(rtab, Wsv + (size_t)i * DKH * DKH, bsv + i * DKH, svT);

            attention_kernel<<<attnGrid, gemmBlk, 0, stream>>>(qb, kb, vb, skT, svT, rl, msk, cx);

            gemm_bf16_kernel<DMODEL, DMODEL, false, true>
                <<<gD, gemmBlk, 0, stream>>>(cx, Wo_i, bo + i * DMODEL, x, x);

            layernorm_kernel<<<MROWS, 256, 0, stream>>>(x, ln2g + i * DMODEL, ln2b + i * DMODEL, h);

            gemm_bf16_kernel<DFFN, DMODEL, true, false>
                <<<gF, gemmBlk, 0, stream>>>(h, W1_i, b1 + i * DFFN, nullptr, ffn);
            gemm_bf16_kernel<DMODEL, DFFN, false, true>
                <<<gD, gemmBlk, 0, stream>>>(ffn, W2_i, b2 + i * DMODEL, x, x);
        }

        layernorm_kernel<<<MROWS, 256, 0, stream>>>(x, lnfg, lnfb, outp);
    }
}
